// MultiHeadAttention_51075751084249
// MI455X (gfx1250) — matrix operand order repaired and checked by reading, not compiled, not run on hardware
//
#include <hip/hip_runtime.h>

// ---------------------------------------------------------------------------
// Fused multi-head attention (column-softmax variant) for MI455X / gfx1250.
//   0) fp32 -> f16 conversion of X, Wqkv, Wo
//   1) QKV GEMM: block computes 128x64, weight tiles double-buffered into LDS
//      via global_load_async_to_lds_b128 (ASYNCcnt), wmma f16 -> f32
//   2) column stats pass: per (b,h) column max M[k] / sum-exp Z[k] (wmma)
//   3) P = exp(s - M)/Z, out = P @ V (wmma, P relaid via LDS)
//   4) output projection GEMM, same async-LDS staging, + bias -> fp32 out
// ---------------------------------------------------------------------------

#define B_    2
#define S_    2048
#define E_    1024
#define H_    16
#define DH_   64
#define MTOK  (B_ * S_)     // 4096 tokens
#define NQKV  (3 * E_)      // 3072 qkv features
#define WPAD  40            // padded LDS row (halfwords): 80B, multiple of 16B

typedef __attribute__((ext_vector_type(16))) _Float16 v16h;
typedef __attribute__((ext_vector_type(8)))  _Float16 v8h;
typedef __attribute__((ext_vector_type(4)))  _Float16 v4h;
typedef __attribute__((ext_vector_type(8)))  float    v8f;

// ---- WMMA wrapper ----------------------------------------------------------
__device__ __forceinline__ v8f wmma16(v16h a, v16h b, v8f c) {
  return __builtin_amdgcn_wmma_f32_16x16x32_f16(
      /*neg_a=*/false, a, /*neg_b=*/false, b,
      /*c_mod=*/(short)0, c, /*reuse_a=*/false, /*reuse_b=*/false);
}

// ---- async copy: 16B global -> LDS, tracked by ASYNCcnt --------------------
// LDS offset = low 32 bits of the generic pointer (ISA 10.2 aperture rule).
__device__ __forceinline__ void async_copy16(_Float16* lds_dst,
                                             const _Float16* gsrc) {
  unsigned loff = (unsigned)(uintptr_t)lds_dst;
  asm volatile("global_load_async_to_lds_b128 %0, %1, off"
               :: "v"(loff), "v"((unsigned long long)(uintptr_t)gsrc)
               : "memory");
}
#define WAIT_ASYNCCNT(N) asm volatile("s_wait_asynccnt %0" :: "n"(N) : "memory")

// ---- Fragment loaders (contiguous-k order, same for A and B) --------------------------------
// A fragment: lane holds row m = lane&15; K = half*16 + e, half = lane>>4.
__device__ __forceinline__ v16h load_a_row(const _Float16* prow, int half) {
  v8h lo = *(const v8h*)(prow + half * 16);
  v8h hi = *(const v8h*)(prow + half * 16 + 8);
  v16h a;
#pragma unroll
  for (int e = 0; e < 8; ++e) { a[e] = lo[e]; a[8 + e] = hi[e]; }
  return a;
}

// B fragment: lane holds column n = lane&15, K = (lane>>4)*16 + e.
// 'base' is n-major / K-contiguous: value(k,n) = base[n*ld + k].
__device__ __forceinline__ v16h load_b16(const _Float16* base, int ld,
                                         int n, int khalf) {
  return *(const v16h*)(base + (size_t)n * ld + khalf * 16);
}

// ---- 0) fp32 -> f16 conversion --------------------------------------------
__global__ __launch_bounds__(256) void cvt_f32_f16(const float* __restrict__ src,
                                                   _Float16* __restrict__ dst,
                                                   int n4) {
  int i = blockIdx.x * 256 + threadIdx.x;
  if (i < n4) {
    float4 v = ((const float4*)src)[i];
    v4h o = {(_Float16)v.x, (_Float16)v.y, (_Float16)v.z, (_Float16)v.w};
    ((v4h*)dst)[i] = o;
  }
}

// ---- shared helper: stage one 64x32 f16 weight tile into LDS ---------------
// 4KB tile, 256 threads x 16B async copies. rowbase = first feature row.
__device__ __forceinline__ void stage_wtile(_Float16* lbuf /*[64][WPAD]*/,
                                            const _Float16* W, int rowbase,
                                            int kk) {
  const int t = threadIdx.x;
  const int row = t >> 2, chk = t & 3;           // 4 x 16B chunks per row
  async_copy16(lbuf + row * WPAD + chk * 8,
               W + (size_t)(rowbase + row) * E_ + kk + chk * 8);
}

// ---- 1) QKV projection GEMM + scatter -------------------------------------
// Block = 8 waves = 128x64 output tile; waves stack along M and share the
// 64-wide weight group, which is double-buffered in LDS via async copies.
__global__ __launch_bounds__(256) void qkv_gemm(
    const _Float16* __restrict__ X16, const _Float16* __restrict__ W16,
    const float* __restrict__ bias,
    _Float16* __restrict__ qbuf, _Float16* __restrict__ kbuf,
    _Float16* __restrict__ vT) {
  __shared__ __align__(16) _Float16 lds_w[2][64][WPAD];
  const int wave = threadIdx.x >> 5, lane = threadIdx.x & 31;
  const int n = lane & 15, half = lane >> 4;
  const int bm = blockIdx.x / (NQKV / 64);      // 0..31 (128-row block)
  const int ntg = blockIdx.x % (NQKV / 64);     // 0..47 (64-wide features)
  const int m0 = bm * 128 + wave * 16;          // this wave's 16 rows

  stage_wtile(&lds_w[0][0][0], W16, ntg * 64, 0);

  v8f acc[4] = {};
  const _Float16* arow = X16 + (size_t)(m0 + n) * E_;
  int ib = 0;
  for (int kk = 0; kk < E_; kk += 32) {
    if (kk + 32 < E_) {
      stage_wtile(&lds_w[ib ^ 1][0][0], W16, ntg * 64, kk + 32);
      WAIT_ASYNCCNT(1);                  // stage kk landed (in-order)
    } else {
      WAIT_ASYNCCNT(0);
    }
    v16h a = load_a_row(arow + kk, half);  // global-only: overlap with barrier
    __syncthreads();                       // publish tile[ib] to all waves
    v16h bf[4];
#pragma unroll
    for (int t = 0; t < 4; ++t)            // issue all ds loads first
      bf[t] = load_b16(&lds_w[ib][t * 16][0], WPAD, n, half);
#pragma unroll
    for (int t = 0; t < 4; ++t)            // then back-to-back wmma
      acc[t] = wmma16(a, bf[t], acc[t]);
    __syncthreads();                       // done reading tile[ib]
    ib ^= 1;
  }

  const int f0 = ntg * 64;
  const int which = f0 >> 10;             // 0=q 1=k 2=v (uniform)
  const int h = (f0 & (E_ - 1)) >> 6;     // head (uniform)
  const int b_idx = m0 >> 11;             // batch (tile never crosses)
#pragma unroll
  for (int t = 0; t < 4; ++t) {
    const int d = (f0 + t * 16 + n) & 63;
    const float bv = bias[f0 + t * 16 + n];
#pragma unroll
    for (int r = 0; r < 8; ++r) {
      const int s = (m0 + r + 8 * half) & (S_ - 1);
      const float val = acc[t][r] + bv;
      const size_t bh = (size_t)b_idx * H_ + h;
      if (which == 0)
        qbuf[(bh * S_ + s) * DH_ + d] = (_Float16)(val * 0.125f); // 1/sqrt(dh)
      else if (which == 1)
        kbuf[(bh * S_ + s) * DH_ + d] = (_Float16)val;
      else
        vT[(bh * DH_ + d) * S_ + s] = (_Float16)val;              // transposed
    }
  }
}

// ---- 2) column softmax stats: M[k] = max_q s, Z[k] = sum_q exp(s-M) -------
__global__ __launch_bounds__(256) void col_stats(
    const _Float16* __restrict__ qbuf, const _Float16* __restrict__ kbuf,
    float* __restrict__ Mbuf, float* __restrict__ Zbuf) {
  const int wave = threadIdx.x >> 5, lane = threadIdx.x & 31;
  const int n = lane & 15, half = lane >> 4;
  const int bh = blockIdx.x >> 4;
  const int col0 = (blockIdx.x & 15) * 128 + wave * 16;

  const _Float16* kbase = kbuf + ((size_t)bh * S_ + col0) * DH_;
  v16h b0 = load_b16(kbase, DH_, n, half);
  v16h b1 = load_b16(kbase + 32, DH_, n, half);

  float m = -3.0e38f, z = 0.0f;
  for (int qb = 0; qb < S_; qb += 16) {
    const _Float16* arow = qbuf + ((size_t)bh * S_ + qb + n) * DH_;
    v16h a0 = load_a_row(arow, half);
    v16h a1 = load_a_row(arow + 32, half);
    v8f c = {};
    c = wmma16(a0, b0, c);
    c = wmma16(a1, b1, c);
    float tmax = c[0];
#pragma unroll
    for (int r = 1; r < 8; ++r) tmax = fmaxf(tmax, c[r]);
    const float mn = fmaxf(m, tmax);
    const float rescale = __expf(m - mn);
    float sl = 0.0f;
#pragma unroll
    for (int r = 0; r < 8; ++r) sl += __expf(c[r] - mn);
    z = z * rescale + sl;
    m = mn;
  }
  const float mo = __shfl_xor(m, 16, 32);
  const float zo = __shfl_xor(z, 16, 32);
  const float mm = fmaxf(m, mo);
  const float zz = z * __expf(m - mm) + zo * __expf(mo - mm);
  if (half == 0) {
    Mbuf[(size_t)bh * S_ + col0 + n] = mm;
    Zbuf[(size_t)bh * S_ + col0 + n] = zz;
  }
}

// ---- 3) out = (exp(s - M)/Z) @ V ------------------------------------------
__global__ __launch_bounds__(256) void attn_pv(
    const _Float16* __restrict__ qbuf, const _Float16* __restrict__ kbuf,
    const _Float16* __restrict__ vT, const float* __restrict__ Mbuf,
    const float* __restrict__ Zbuf, _Float16* __restrict__ aout) {
  __shared__ __align__(16) _Float16 lds_p[8][16][32];
  const int wave = threadIdx.x >> 5, lane = threadIdx.x & 31;
  const int n = lane & 15, half = lane >> 4;
  const int bh = blockIdx.x >> 4;
  const int qb = (blockIdx.x & 15) * 128 + wave * 16;

  const _Float16* arow = qbuf + ((size_t)bh * S_ + qb + n) * DH_;
  v16h a0 = load_a_row(arow, half);
  v16h a1 = load_a_row(arow + 32, half);

  v8f acc[4] = {};
  for (int kb = 0; kb < S_; kb += 32) {
    // prefetch next key block (128B cachelines, spread over lanes)
    if (kb + 32 < S_) {
      const char* pf = (const char*)(kbuf + ((size_t)bh * S_ + kb + 32) * DH_);
      __builtin_prefetch(pf + lane * 128, 0, 1);
    }
#pragma unroll
    for (int t = 0; t < 2; ++t) {
      const int col0 = kb + t * 16;
      const _Float16* kbase = kbuf + ((size_t)bh * S_ + col0) * DH_;
      v16h b0 = load_b16(kbase, DH_, n, half);
      v16h b1 = load_b16(kbase + 32, DH_, n, half);
      v8f c = {};
      c = wmma16(a0, b0, c);
      c = wmma16(a1, b1, c);
      const float Mv = Mbuf[(size_t)bh * S_ + col0 + n];
      const float Zr = 1.0f / Zbuf[(size_t)bh * S_ + col0 + n];
#pragma unroll
      for (int r = 0; r < 8; ++r)
        lds_p[wave][r + 8 * half][t * 16 + n] =
            (_Float16)(__expf(c[r] - Mv) * Zr);
    }
    // issue all V-fragment loads, then reload P, then back-to-back wmma
    v16h vb[4];
#pragma unroll
    for (int db = 0; db < 4; ++db)
      vb[db] = load_b16(vT + ((size_t)bh * DH_ + db * 16) * S_ + kb, S_, n, half);
    v16h pa = load_a_row(&lds_p[wave][n][0], half);
#pragma unroll
    for (int db = 0; db < 4; ++db)
      acc[db] = wmma16(pa, vb[db], acc[db]);
  }

  const int b_idx = bh >> 4, h = bh & 15;
#pragma unroll
  for (int db = 0; db < 4; ++db)
#pragma unroll
    for (int r = 0; r < 8; ++r) {
      const int s = qb + r + 8 * half;
      aout[((size_t)(b_idx * S_ + s)) * E_ + h * DH_ + db * 16 + n] =
          (_Float16)acc[db][r];
    }
}

// ---- 4) output projection: out = aout @ Wo^T + bo (fp32 result) -----------
__global__ __launch_bounds__(256) void out_proj(
    const _Float16* __restrict__ aout, const _Float16* __restrict__ Wo16,
    const float* __restrict__ bo, float* __restrict__ out) {
  __shared__ __align__(16) _Float16 lds_w[2][64][WPAD];
  const int wave = threadIdx.x >> 5, lane = threadIdx.x & 31;
  const int n = lane & 15, half = lane >> 4;
  const int bm = blockIdx.x / (E_ / 64);        // 0..31
  const int ntg = blockIdx.x % (E_ / 64);       // 0..15
  const int m0 = bm * 128 + wave * 16;

  stage_wtile(&lds_w[0][0][0], Wo16, ntg * 64, 0);

  v8f acc[4] = {};
  const _Float16* arow = aout + (size_t)(m0 + n) * E_;
  int ib = 0;
  for (int kk = 0; kk < E_; kk += 32) {
    if (kk + 32 < E_) {
      stage_wtile(&lds_w[ib ^ 1][0][0], Wo16, ntg * 64, kk + 32);
      WAIT_ASYNCCNT(1);
    } else {
      WAIT_ASYNCCNT(0);
    }
    v16h a = load_a_row(arow + kk, half);
    __syncthreads();
    v16h bf[4];
#pragma unroll
    for (int t = 0; t < 4; ++t)
      bf[t] = load_b16(&lds_w[ib][t * 16][0], WPAD, n, half);
#pragma unroll
    for (int t = 0; t < 4; ++t)
      acc[t] = wmma16(a, bf[t], acc[t]);
    __syncthreads();
    ib ^= 1;
  }
#pragma unroll
  for (int t = 0; t < 4; ++t) {
    const float bv = bo[ntg * 64 + t * 16 + n];
#pragma unroll
    for (int r = 0; r < 8; ++r)
      out[(size_t)(m0 + r + 8 * half) * E_ + ntg * 64 + t * 16 + n] =
          acc[t][r] + bv;
  }
}

// ---------------------------------------------------------------------------
extern "C" void kernel_launch(void* const* d_in, const int* in_sizes, int n_in,
                              void* d_out, int out_size, void* d_ws,
                              size_t ws_size, hipStream_t stream) {
  (void)in_sizes; (void)n_in; (void)out_size; (void)ws_size;
  const float* X    = (const float*)d_in[0];   // [B,S,E]
  const float* Wqkv = (const float*)d_in[1];   // [3E,E]
  const float* bqkv = (const float*)d_in[2];   // [3E]
  const float* Wo   = (const float*)d_in[3];   // [E,E]
  const float* bo   = (const float*)d_in[4];   // [E]
  float* out = (float*)d_out;

  char* p = (char*)d_ws;
  _Float16* X16  = (_Float16*)p; p += (size_t)MTOK * E_ * 2;          // 8 MB
  _Float16* Wq16 = (_Float16*)p; p += (size_t)NQKV * E_ * 2;          // 6 MB
  _Float16* Wo16 = (_Float16*)p; p += (size_t)E_ * E_ * 2;            // 2 MB
  _Float16* qbuf = (_Float16*)p; p += (size_t)B_ * H_ * S_ * DH_ * 2; // 8 MB
  _Float16* kbuf = (_Float16*)p; p += (size_t)B_ * H_ * S_ * DH_ * 2; // 8 MB
  _Float16* vT   = (_Float16*)p; p += (size_t)B_ * H_ * S_ * DH_ * 2; // 8 MB
  float* Mbuf    = (float*)p;    p += (size_t)B_ * H_ * S_ * 4;       // 256 KB
  float* Zbuf    = (float*)p;    p += (size_t)B_ * H_ * S_ * 4;       // 256 KB
  _Float16* aout = (_Float16*)p;                                      // 8 MB

  cvt_f32_f16<<<MTOK * E_ / 4 / 256, 256, 0, stream>>>(X, X16, MTOK * E_ / 4);
  cvt_f32_f16<<<NQKV * E_ / 4 / 256, 256, 0, stream>>>(Wqkv, Wq16, NQKV * E_ / 4);
  cvt_f32_f16<<<E_ * E_ / 4 / 256, 256, 0, stream>>>(Wo, Wo16, E_ * E_ / 4);

  qkv_gemm<<<(MTOK / 128) * (NQKV / 64), 256, 0, stream>>>(
      X16, Wq16, bqkv, qbuf, kbuf, vT);                         // 1536 blocks
  col_stats<<<B_ * H_ * (S_ / 128), 256, 0, stream>>>(
      qbuf, kbuf, Mbuf, Zbuf);                                  // 512 blocks
  attn_pv<<<B_ * H_ * (S_ / 128), 256, 0, stream>>>(
      qbuf, kbuf, vT, Mbuf, Zbuf, aout);                        // 512 blocks
  out_proj<<<(MTOK / 128) * (E_ / 64), 256, 0, stream>>>(
      aout, Wo16, bo, out);                                     // 512 blocks
}
